// Attention_16690242912429
// MI455X (gfx1250) — matrix operand order repaired and checked by reading, not compiled, not run on hardware
//
#include <hip/hip_runtime.h>
#include <math.h>

#define T_  128
#define N_  1024
#define E_  16384
#define D_  256
#define D2_ 512
#define H_  8
#define HD_ 32
#define NN_ (N_*N_)

typedef __attribute__((ext_vector_type(16))) __bf16 v16bf;
typedef __attribute__((ext_vector_type(8)))  float  v8f;

union Frag32B {
    uint4    u4[2];
    unsigned u[8];
    v16bf    bf;
};

__device__ __forceinline__ unsigned short f32_to_bf16(float f) {
    unsigned x = __builtin_bit_cast(unsigned, f);
    x += 0x7FFFu + ((x >> 16) & 1u);          // round-to-nearest-even
    return (unsigned short)(x >> 16);
}
__device__ __forceinline__ float bf16_to_f32(unsigned short h) {
    return __builtin_bit_cast(float, ((unsigned)h) << 16);
}

// ---------------------------------------------------------------- scatter A
__global__ __launch_bounds__(256)
void k_scatter(const float* __restrict__ ev, const int* __restrict__ es,
               const int* __restrict__ ed, float* __restrict__ A) {
    long long gid = (long long)blockIdx.x * blockDim.x + threadIdx.x;
    if (gid >= (long long)T_ * E_) return;
    int t = (int)(gid >> 14);                 // E = 2^14
    int idx = (int)gid;
    int s = es[idx], d = ed[idx];
    atomicAdd(&A[(size_t)t * NN_ + (size_t)s * N_ + d], ev[idx]);
}

// ------------------------------------------------ embeddings f32 -> bf16
__global__ __launch_bounds__(256)
void k_emb_bf16(const float* __restrict__ emb, unsigned* __restrict__ embh) {
    long long i = (long long)blockIdx.x * 256 + threadIdx.x;   // float4 index
    const float4 v = ((const float4*)emb)[i];
    uint2 o;
    o.x = (unsigned)f32_to_bf16(v.x) | ((unsigned)f32_to_bf16(v.y) << 16);
    o.y = (unsigned)f32_to_bf16(v.z) | ((unsigned)f32_to_bf16(v.w) << 16);
    ((uint2*)embh)[i] = o;
}

// -------- MLP1 + avg/max pool, fast path: grid (T, 2), 512 thr (16 waves).
// Each wave owns one 16-col tile; W1 bf16 B-frags in registers (no spills at
// 512 thr: VGPR cap ~2x vs 1024 thr). bf16 A-tiles double-buffered in LDS via
// global_load_async_to_lds_b128 issued by wave 0, s_wait_asynccnt + barrier.
__global__ __launch_bounds__(512)
void k_mlp1_fast(const unsigned short* __restrict__ embh,
                 const float* __restrict__ W1, const float* __restrict__ b1,
                 float* __restrict__ pooled) {
    __shared__ __align__(16) unsigned short atile[2][16][264];  // 2 x 8448 B
    const int t      = blockIdx.x;
    const int cg     = blockIdx.y * 256;      // column group base
    const int tid    = threadIdx.x;
    const int lane   = tid & 31;
    const int wave   = tid >> 5;              // 0..15 -> column tile
    const int m      = lane & 15;
    const int khalfA = (lane >> 4) * 16;
    const int col    = cg + wave * 16 + (lane & 15);

    // preload this wave's W1 column-tile as packed-bf16 B fragments
    unsigned bfrag[8][8];
#pragma unroll
    for (int kc = 0; kc < 8; ++kc) {
        const int kb = kc * 32 + (lane >> 4) * 16;
#pragma unroll
        for (int i = 0; i < 8; ++i) {
            float p0 = W1[(size_t)(kb + 2 * i)     * D2_ + col];
            float p1 = W1[(size_t)(kb + 2 * i + 1) * D2_ + col];
            bfrag[kc][i] = (unsigned)f32_to_bf16(p0) |
                           ((unsigned)f32_to_bf16(p1) << 16);
        }
    }
    const float bias = b1[col];
    const unsigned short* src = embh + (size_t)t * N_ * D_;

    auto issue = [&](int nc, int buf) {       // wave 0 only: 16 rows x 512 B
        const unsigned short* g0 = src + (size_t)nc * 16 * D_;
#pragma unroll
        for (int r = 0; r < 16; ++r) {
            unsigned ldsoff = (unsigned)(size_t)&atile[buf][r][0] + lane * 16;
            const char* ga  = (const char*)(g0 + r * D_) + lane * 16;
            asm volatile("global_load_async_to_lds_b128 %0, %1, off"
                         :: "v"(ldsoff), "v"(ga) : "memory");
        }
    };

    float sumc[8], maxc[8];
#pragma unroll
    for (int i = 0; i < 8; ++i) { sumc[i] = 0.f; maxc[i] = 0.f; } // relu >= 0

    if (wave == 0) issue(0, 0);
    for (int nc = 0; nc < N_ / 16; ++nc) {
        const int cur = nc & 1;
        if (wave == 0)
            asm volatile("s_wait_asynccnt 0x0" ::: "memory");
        __syncthreads();                              // tile `cur` published
        if (wave == 0 && nc + 1 < N_ / 16) issue(nc + 1, cur ^ 1);

        v8f c = {};
#pragma unroll
        for (int kc = 0; kc < 8; ++kc) {
            const int kb = kc * 32;
            Frag32B a, b;
            a.u4[0] = *(const uint4*)&atile[cur][m][kb + khalfA];
            a.u4[1] = *(const uint4*)&atile[cur][m][kb + khalfA + 8];
#pragma unroll
            for (int i = 0; i < 8; ++i) b.u[i] = bfrag[kc][i];
            c = __builtin_amdgcn_wmma_f32_16x16x32_bf16(
                    false, a.bf, false, b.bf, (short)0, c, false, false);
        }
#pragma unroll
        for (int i = 0; i < 8; ++i) {
            float h = fmaxf(c[i] + bias, 0.f);
            sumc[i] += h;
            maxc[i] = fmaxf(maxc[i], h);
        }
        __syncthreads();                 // all reads of `cur` done before refill
    }
    float s = 0.f, mx = 0.f;
#pragma unroll
    for (int i = 0; i < 8; ++i) { s += sumc[i]; mx = fmaxf(mx, maxc[i]); }
    s  += __shfl_xor(s, 16, 32);
    mx  = fmaxf(mx, __shfl_xor(mx, 16, 32));
    if (lane < 16)
        pooled[t * D2_ + col] = s * (1.0f / N_) + mx;
}

// ---------------- MLP1 fallback (small ws): grid (T,8), 128 thr, f32 source
__global__ __launch_bounds__(128)
void k_mlp1_pool(const float* __restrict__ emb, const float* __restrict__ W1,
                 const float* __restrict__ b1, float* __restrict__ pooled) {
    __shared__ __align__(16) unsigned short w1t[64][264];   // [col][k] bf16
    __shared__ __align__(16) unsigned short atile[16][264]; // [n][k]  bf16
    const int t    = blockIdx.x;
    const int cg   = blockIdx.y * 64;
    const int tid  = threadIdx.x;
    const int lane = tid & 31;
    const int wave = tid >> 5;                // 0..3

    for (int idx = tid; idx < 64 * 256; idx += 128) {
        int col = idx & 63, k = idx >> 6;
        w1t[col][k] = f32_to_bf16(W1[k * D2_ + cg + col]);
    }
    const float bias   = b1[cg + wave * 16 + (lane & 15)];
    const int   m      = lane & 15;
    const int   khalfA = (lane >> 4) * 16;
    const int   khalfB = (lane >> 4) * 16;
    const int   colL   = wave * 16 + (lane & 15);

    float sumc[8], maxc[8];
#pragma unroll
    for (int i = 0; i < 8; ++i) { sumc[i] = 0.f; maxc[i] = 0.f; }

    for (int nc = 0; nc < N_ / 16; ++nc) {
        __syncthreads();
        const float* src = emb + ((size_t)t * N_ + (size_t)nc * 16) * D_;
        for (int idx = tid; idx < 16 * 256; idx += 128) {
            int r = idx >> 8, k = idx & 255;
            atile[r][k] = f32_to_bf16(src[r * D_ + k]);
        }
        __syncthreads();

        v8f c = {};
#pragma unroll
        for (int kc = 0; kc < 8; ++kc) {
            const int kb = kc * 32;
            Frag32B a, b;
            a.u4[0] = *(const uint4*)&atile[m][kb + khalfA];
            a.u4[1] = *(const uint4*)&atile[m][kb + khalfA + 8];
            const uint4* bp = (const uint4*)&w1t[colL][kb + khalfB];
            b.u4[0] = bp[0];
            b.u4[1] = bp[1];
            c = __builtin_amdgcn_wmma_f32_16x16x32_bf16(
                    false, a.bf, false, b.bf, (short)0, c, false, false);
        }
#pragma unroll
        for (int i = 0; i < 8; ++i) {
            float h = fmaxf(c[i] + bias, 0.f);
            sumc[i] += h;
            maxc[i] = fmaxf(maxc[i], h);
        }
    }
    float s = 0.f, mx = 0.f;
#pragma unroll
    for (int i = 0; i < 8; ++i) { s += sumc[i]; mx = fmaxf(mx, maxc[i]); }
    s  += __shfl_xor(s, 16, 32);
    mx  = fmaxf(mx, __shfl_xor(mx, 16, 32));
    if (lane < 16)
        pooled[t * D2_ + cg + wave * 16 + lane] = s * (1.0f / N_) + mx;
}

// ----------------------------------------------------- MLP2 + pos encoding
__global__ __launch_bounds__(256)
void k_mlp2(const float* __restrict__ pooled, const float* __restrict__ W2,
            const float* __restrict__ b2, float* __restrict__ x) {
    const int t = blockIdx.x, d = threadIdx.x;
    float acc = b2[d];
    const float* pr = pooled + t * D2_;
    for (int j = 0; j < D2_; ++j) acc = fmaf(pr[j], W2[j * D_ + d], acc);
    acc = fmaxf(acc, 0.f);
    float div = expf((float)((d >> 1) * 2) * (-9.210340371976184f / 256.0f));
    float ang = (float)t * div;
    float pe  = (d & 1) ? cosf(ang) : sinf(ang);
    x[t * D_ + d] = acc + pe;
}

// --------------------------------------------------------- Q/K projections
__global__ __launch_bounds__(256)
void k_qk(const float* __restrict__ x, const float* __restrict__ Wq,
          const float* __restrict__ bq, const float* __restrict__ Wk,
          const float* __restrict__ bk, float* __restrict__ q,
          float* __restrict__ k) {
    const int t = blockIdx.x, d = threadIdx.x;
    float qa = bq[d], ka = bk[d];
    const float* xr = x + t * D_;
    for (int j = 0; j < D_; ++j) {
        float xv = xr[j];
        qa = fmaf(xv, Wq[j * D_ + d], qa);
        ka = fmaf(xv, Wk[j * D_ + d], ka);
    }
    q[t * D_ + d] = qa * 0.17677669529663687f;  // hd^-0.5, hd=32
    k[t * D_ + d] = ka;
}

// --------------- scores -> softmax -> head-mean -> threshold/eye/tril mask
__global__ __launch_bounds__(256)
void k_probs(const float* __restrict__ q, const float* __restrict__ k,
             float* __restrict__ probs) {
    __shared__ float sc[H_][T_];
    __shared__ float hmax[H_], hsum[H_];
    const int tq = blockIdx.x, tid = threadIdx.x;
    for (int idx = tid; idx < H_ * T_; idx += 256) {
        int h = idx >> 7, tk = idx & 127;
        const float* qr = q + tq * D_ + h * HD_;
        const float* kr = k + tk * D_ + h * HD_;
        float s = 0.f;
#pragma unroll
        for (int i = 0; i < HD_; ++i) s = fmaf(qr[i], kr[i], s);
        sc[h][tk] = s;
    }
    __syncthreads();
    if (tid < H_) {
        float m = -1e30f;
        for (int tk = 0; tk < T_; ++tk) m = fmaxf(m, sc[tid][tk]);
        float ss = 0.f;
        for (int tk = 0; tk < T_; ++tk) ss += expf(sc[tid][tk] - m);
        hmax[tid] = m; hsum[tid] = ss;
    }
    __syncthreads();
    if (tid < T_) {
        int tk = tid;
        float p = 0.f;
#pragma unroll
        for (int h = 0; h < H_; ++h) p += expf(sc[h][tk] - hmax[h]) / hsum[h];
        p *= (1.0f / H_);
        if (p < (1.0f / T_)) p = 0.f;
        if (tk == tq) p += 1.f;
        if (tk > tq)  p = 0.f;
        probs[tq * T_ + tk] = p;
    }
}

// ------------- out[t, c] = sum_s probs[t,s] * A[s, c]  (in-place in d_out)
__global__ __launch_bounds__(256)
void k_adjmix(const float* __restrict__ probs, float* __restrict__ Aout) {
    __shared__ __align__(16) unsigned short Ah[64][136];  // [col][s] hi
    __shared__ __align__(16) unsigned short Al[64][136];  // [col][s] lo
    const int    tid  = threadIdx.x;
    const int    lane = tid & 31;
    const int    wave = tid >> 5;             // mtile 0..7 -> t rows
    const size_t cb   = (size_t)blockIdx.x * 64;

    const int m     = lane & 15;
    const int khalf = (lane >> 4) * 16;
    const int trow  = wave * 16 + m;
    Frag32B Ph[4], Pl[4];
#pragma unroll
    for (int kc = 0; kc < 4; ++kc) {
        const float* pr = probs + trow * T_ + kc * 32 + khalf;
#pragma unroll
        for (int g = 0; g < 2; ++g) {
#pragma unroll
            for (int jp = 0; jp < 4; ++jp) {
                float p0 = pr[g * 8 + jp * 2];
                float p1 = pr[g * 8 + jp * 2 + 1];
                unsigned short h0 = f32_to_bf16(p0), h1 = f32_to_bf16(p1);
                float l0 = p0 - bf16_to_f32(h0), l1 = p1 - bf16_to_f32(h1);
                Ph[kc].u[g * 4 + jp] = (unsigned)h0 | ((unsigned)h1 << 16);
                Pl[kc].u[g * 4 + jp] =
                    (unsigned)f32_to_bf16(l0) | ((unsigned)f32_to_bf16(l1) << 16);
            }
        }
    }

    for (int idx = tid; idx < 128 * 64; idx += 256) {
        int col = idx & 63, s = idx >> 6;
        float v = Aout[(size_t)s * NN_ + cb + col];
        unsigned short h = f32_to_bf16(v);
        Ah[col][s] = h;
        Al[col][s] = f32_to_bf16(v - bf16_to_f32(h));
    }
    __syncthreads();   // strip fully captured -> in-place writes now safe

    const int khalfB = (lane >> 4) * 16;
#pragma unroll
    for (int nt = 0; nt < 4; ++nt) {
        const int colL = nt * 16 + (lane & 15);
        v8f c = {};
#pragma unroll
        for (int kc = 0; kc < 4; ++kc) {
            Frag32B bh, bl;
            const uint4* ph = (const uint4*)&Ah[colL][kc * 32 + khalfB];
            bh.u4[0] = ph[0]; bh.u4[1] = ph[1];
            const uint4* pl = (const uint4*)&Al[colL][kc * 32 + khalfB];
            bl.u4[0] = pl[0]; bl.u4[1] = pl[1];
            c = __builtin_amdgcn_wmma_f32_16x16x32_bf16(
                    false, Ph[kc].bf, false, bh.bf, (short)0, c, false, false);
            c = __builtin_amdgcn_wmma_f32_16x16x32_bf16(
                    false, Pl[kc].bf, false, bh.bf, (short)0, c, false, false);
            c = __builtin_amdgcn_wmma_f32_16x16x32_bf16(
                    false, Ph[kc].bf, false, bl.bf, (short)0, c, false, false);
        }
        const size_t colG = cb + nt * 16 + (lane & 15);
#pragma unroll
        for (int i = 0; i < 8; ++i) {
            int tr = wave * 16 + (lane >> 4) * 8 + i;
            Aout[(size_t)tr * NN_ + colG] = c[i];
        }
    }
}

// --------------------------------------------------------------------------
extern "C" void kernel_launch(void* const* d_in, const int* in_sizes, int n_in,
                              void* d_out, int out_size, void* d_ws, size_t ws_size,
                              hipStream_t stream) {
    const float* emb = (const float*)d_in[0];
    const float* W1  = (const float*)d_in[1];
    const float* b1  = (const float*)d_in[2];
    const float* W2  = (const float*)d_in[3];
    const float* b2  = (const float*)d_in[4];
    const float* Wq  = (const float*)d_in[5];
    const float* bq  = (const float*)d_in[6];
    const float* Wk  = (const float*)d_in[7];
    const float* bk  = (const float*)d_in[8];
    const float* ev  = (const float*)d_in[9];
    const int*   es  = (const int*)d_in[10];
    const int*   ed  = (const int*)d_in[11];
    float* out = (float*)d_out;
    (void)in_sizes; (void)n_in; (void)out_size;

    const size_t embh_bytes  = (size_t)T_ * N_ * D_ * 2;              // 64 MiB
    const size_t tail_floats = (size_t)T_ * D2_ + 3u * T_ * D_ + T_ * T_;
    const bool   fast        = ws_size >= embh_bytes + tail_floats * 4;

    float* fls = fast ? (float*)((char*)d_ws + embh_bytes) : (float*)d_ws;
    float* pooled = fls;                     // T*2D
    float* x      = pooled + T_ * D2_;       // T*D
    float* q      = x + T_ * D_;             // T*D
    float* k      = q + T_ * D_;             // T*D
    float* probs  = k + T_ * D_;             // T*T

    // A is built in-place inside d_out, then overwritten strip-by-strip.
    hipMemsetAsync(out, 0, (size_t)T_ * NN_ * sizeof(float), stream);
    k_scatter<<<(T_ * E_) / 256, 256, 0, stream>>>(ev, es, ed, out);

    if (fast) {
        unsigned* embh = (unsigned*)d_ws;
        k_emb_bf16<<<(T_ * N_ * D_ / 4) / 256, 256, 0, stream>>>(emb, embh);
        k_mlp1_fast<<<dim3(T_, 2), 512, 0, stream>>>((const unsigned short*)embh,
                                                     W1, b1, pooled);
    } else {
        k_mlp1_pool<<<dim3(T_, 8), 128, 0, stream>>>(emb, W1, b1, pooled);
    }
    k_mlp2<<<T_, D_, 0, stream>>>(pooled, W2, b2, x);
    k_qk<<<T_, D_, 0, stream>>>(x, Wq, bq, Wk, bk, q, k);
    k_probs<<<T_, 256, 0, stream>>>(q, k, probs);
    k_adjmix<<<NN_ / 64, 256, 0, stream>>>(probs, out);
}
